// SimpleLSTMTop_73727408603248
// MI455X (gfx1250) — matrix operand order repaired and checked by reading, not compiled, not run on hardware
//
#include <hip/hip_runtime.h>
#include <stdint.h>

typedef __attribute__((ext_vector_type(16))) __bf16 v16bf;
typedef __attribute__((ext_vector_type(8)))  float  v8f;

union V16 { v16bf v; uint4 q[2]; unsigned u[8]; };

#define XS 260                    // staged x row stride (f32): 256 + 4 pad (bank + align)
#define HS 136                    // staged h row stride (bf16): 128 + 8 pad
#define XROWB (XS * 4)            // 1040 B per staged x row
#define XBUF_BYTES (16 * XS * 4)  // 16640
#define HBUF_BYTES (16 * HS * 2)  // 4352
#define WAVE_BYTES (2 * XBUF_BYTES + HBUF_BYTES)
#define WAVES_PER_WG 8
#define WG_SIZE (WAVES_PER_WG * 32)
#define SMEM_BYTES (WAVE_BYTES * WAVES_PER_WG)  // 301056 B <= 320KB WGP LDS
#define NWG 256

static_assert(SMEM_BYTES <= 320 * 1024, "LDS budget");

// Pre-swizzled weight-tile arena offsets (bf16 elements). Each tile is
// 32 lanes x 16 elems = 512 bf16 = 1KB, laid out exactly as the WMMA
// B-matrix (32x16, K x N) VGPR layout: elem e of lane l = B[(l/16)*8 + (e<8 ? e : e+8)][l%16].
#define WT_L1 0       // 8x8 tiles (K=256,N=128)
#define WT_L2 32768   // 4x4
#define WT_L3 40960   // 2x2
#define WT_L4 43008   // 1x1 (K=32,N=16)
#define WT_L5 43520   // 1x1 (K=16->32, N=8->16, zero padded)
#define WT_L6 44032   // 1x1 (K=8->32,  N=4->16)
#define WT_L7 44544   // 1x1 (K=4->32,  N=1->16)
#define WT_TOTAL 45056

// ---- bf16 conversion helpers -----------------------------------------------
// Pack two f32 -> two bf16 in one dword: round-to-nearest via +0x8000, then a
// single v_perm_b32 grabs the two high halves. 3 VALU per 2 elements.
__device__ __forceinline__ unsigned pkbf(float lo, float hi) {
  unsigned a = __float_as_uint(lo) + 0x8000u;
  unsigned b = __float_as_uint(hi) + 0x8000u;
  // bytes {b3,b2,a3,a2} -> {hi16(b), hi16(a)}
  return __builtin_amdgcn_perm(b, a, 0x07060302u);
}
__device__ __forceinline__ unsigned short bf1(float a) {  // prep kernel only
  unsigned u = __float_as_uint(a);
  u += 0x7FFFu + ((u >> 16) & 1u);
  return (unsigned short)(u >> 16);
}

// ---------------------------------------------------------------------------
// Prep: convert fp32 weights -> bf16 tiles in B-matrix lane layout, zero-pad.
// ---------------------------------------------------------------------------
__global__ void prep_wtiles(const float* __restrict__ W1, const float* __restrict__ W2,
                            const float* __restrict__ W3, const float* __restrict__ W4,
                            const float* __restrict__ W5, const float* __restrict__ W6,
                            const float* __restrict__ W7, unsigned short* __restrict__ wt) {
  const int layerTiles[7] = {64, 16, 4, 1, 1, 1, 1};
  const int layerBase[7]  = {WT_L1, WT_L2, WT_L3, WT_L4, WT_L5, WT_L6, WT_L7};
  const int NTn[7]        = {8, 4, 2, 1, 1, 1, 1};
  const int Kr[7]         = {256, 128, 64, 32, 16, 8, 4};
  const int Nr[7]         = {128, 64, 32, 16, 8, 4, 1};
  const float* Ws[7] = {W1, W2, W3, W4, W5, W6, W7};

  int t = blockIdx.x, layer = 0, acc = 0;
  while (layer < 7 && t >= acc + layerTiles[layer]) { acc += layerTiles[layer]; ++layer; }
  if (layer >= 7) return;
  const int tl = t - acc;
  const int kt = tl / NTn[layer], nt = tl % NTn[layer];
  const int lane = threadIdx.x & 31;
  const int K = Kr[layer], N = Nr[layer];
  const float* W = Ws[layer];
  unsigned short* dst = wt + layerBase[layer] + tl * 512 + lane * 16;
#pragma unroll
  for (int e = 0; e < 16; ++e) {
    int k = kt * 32 + (lane >> 4) * 8 + (e < 8 ? e : e + 8);   // B-matrix K index for this slot
    int n = nt * 16 + (lane & 15);            // B-matrix N index for this slot
    float v = (k < K && n < N) ? W[k * N + n] : 0.0f;
    dst[e] = bf1(v);
  }
}

// ---------------------------------------------------------------------------
// Async x-tile prefetch: GVS form, SGPR base + incrementally updated VGPR
// offsets. Global side is contiguous (idx*16B); LDS side has +16B row pad.
// ---------------------------------------------------------------------------
__device__ __forceinline__ void prefetch_x_tile(const float* __restrict__ x, int tile,
                                                unsigned lds_base, int lane) {
  const float* gp = x + (size_t)tile * (16 * 256);   // wave-uniform
  unsigned laA = lds_base + lane * 16;               // LDS addr, first half-row
  unsigned laB = laA + 512;                          // LDS addr, second half-row
  unsigned vo  = lane * 16;                          // global byte offset
#pragma unroll
  for (int r = 0; r < 16; ++r) {                     // one 256-f32 row per iter
    asm volatile("global_load_async_to_lds_b128 %0, %1, %2 th:TH_LOAD_NT"
                 :: "v"(laA), "v"(vo), "s"(gp) : "memory");
    vo += 512;
    asm volatile("global_load_async_to_lds_b128 %0, %1, %2 th:TH_LOAD_NT"
                 :: "v"(laB), "v"(vo), "s"(gp) : "memory");
    vo += 512;
    laA += XROWB;
    laB += XROWB;
  }
}

// Initialize accumulators with the bias (WMMA computes A*B + C).
template <int NTL>
__device__ __forceinline__ void init_bias(v8f* acc, const float* __restrict__ bias,
                                          int realN, int lane) {
  const int n0 = lane & 15;
#pragma unroll
  for (int nt = 0; nt < NTL; ++nt) {
    const int n = nt * 16 + n0;
    const float bv = (n < realN) ? bias[n] : 0.0f;
#pragma unroll
    for (int v = 0; v < 8; ++v) acc[nt][v] = bv;
  }
}

// Layer 1: A gathered from f32 LDS staging with on-the-fly bf16 convert.
__device__ __forceinline__ void gemm_l1(const float* xrow, const unsigned short* __restrict__ wt,
                                        v8f* acc, int lane) {
  const int h = lane >> 4;
#pragma unroll
  for (int kt = 0; kt < 8; ++kt) {
    const int k0 = kt * 32 + h * 8;
    float4 f0 = *(const float4*)(xrow + k0);
    float4 f1 = *(const float4*)(xrow + k0 + 4);
    float4 f2 = *(const float4*)(xrow + k0 + 16);
    float4 f3 = *(const float4*)(xrow + k0 + 20);
    V16 a;
    a.u[0] = pkbf(f0.x, f0.y); a.u[1] = pkbf(f0.z, f0.w);
    a.u[2] = pkbf(f1.x, f1.y); a.u[3] = pkbf(f1.z, f1.w);
    a.u[4] = pkbf(f2.x, f2.y); a.u[5] = pkbf(f2.z, f2.w);
    a.u[6] = pkbf(f3.x, f3.y); a.u[7] = pkbf(f3.z, f3.w);
#pragma unroll
    for (int nt = 0; nt < 8; ++nt) {
      V16 b;
      const uint4* tp = (const uint4*)(wt + (kt * 8 + nt) * 512 + lane * 16);
      b.q[0] = tp[0]; b.q[1] = tp[1];
      acc[nt] = __builtin_amdgcn_wmma_f32_16x16x32_bf16(
          false, a.v, false, b.v, (short)0, acc[nt], false, false);
    }
  }
}

// Layers 2+: A gathered from bf16 LDS staging.
template <int KT, int NTL>
__device__ __forceinline__ void gemm_bf16(const unsigned short* hsrc,
                                          const unsigned short* __restrict__ wt,
                                          v8f* acc, int lane) {
  const int h = lane >> 4;
  const int m = lane & 15;
  const unsigned short* row = hsrc + m * HS;
#pragma unroll
  for (int kt = 0; kt < KT; ++kt) {
    const int k0 = kt * 32 + h * 8;
    V16 a;
    a.q[0] = *(const uint4*)(row + k0);
    a.q[1] = *(const uint4*)(row + k0 + 16);
#pragma unroll
    for (int nt = 0; nt < NTL; ++nt) {
      V16 b;
      const uint4* tp = (const uint4*)(wt + (kt * NTL + nt) * 512 + lane * 16);
      b.q[0] = tp[0]; b.q[1] = tp[1];
      acc[nt] = __builtin_amdgcn_wmma_f32_16x16x32_bf16(
          false, a.v, false, b.v, (short)0, acc[nt], false, false);
    }
  }
}

// ReLU + bf16 convert, write activations back to LDS h-buffer (C/D layout:
// VGPR v of lane h*16+c holds D[v + 8h][c]). Row pairs share one v_perm pack;
// the two halves go to adjacent rows (high half can use ds_store_b16_d16_hi).
template <int NTL>
__device__ __forceinline__ void epilogue_relu(const v8f* acc, unsigned short* hdst, int lane) {
  const int h = lane >> 4, n0 = lane & 15;
  unsigned short* base = hdst + (8 * h) * HS + n0;
#pragma unroll
  for (int nt = 0; nt < NTL; ++nt) {
#pragma unroll
    for (int v = 0; v < 8; v += 2) {
      float t0 = fmaxf(acc[nt][v], 0.0f);
      float t1 = fmaxf(acc[nt][v + 1], 0.0f);
      unsigned u = pkbf(t0, t1);
      base[v * HS + nt * 16]       = (unsigned short)u;
      base[(v + 1) * HS + nt * 16] = (unsigned short)(u >> 16);
    }
  }
}

// ---------------------------------------------------------------------------
// Fused MLP: per wave, 16 batch rows per iteration, double-buffered async x.
// ---------------------------------------------------------------------------
__global__ __launch_bounds__(WG_SIZE, 1) void mlp_fused(
    const float* __restrict__ x, const unsigned short* __restrict__ wt,
    const float* __restrict__ b1, const float* __restrict__ b2,
    const float* __restrict__ b3, const float* __restrict__ b4,
    const float* __restrict__ b5, const float* __restrict__ b6,
    const float* __restrict__ b7, float* __restrict__ out, int ntiles) {
  extern __shared__ char smem[];
  const int lane = threadIdx.x & 31;
  const int wid  = threadIdx.x >> 5;
  const int h    = lane >> 4;
  const int m    = lane & 15;

  char* wavemem = smem + wid * WAVE_BYTES;
  float* xbuf0 = (float*)wavemem;
  float* xbuf1 = (float*)(wavemem + XBUF_BYTES);
  unsigned short* hbuf = (unsigned short*)(wavemem + 2 * XBUF_BYTES);
  unsigned xlds[2];
  xlds[0] = (unsigned)(unsigned long long)(uintptr_t)(void*)xbuf0;  // LDS byte offset
  xlds[1] = (unsigned)(unsigned long long)(uintptr_t)(void*)xbuf1;

  const int gwave  = blockIdx.x * WAVES_PER_WG + wid;
  const int nwaves = gridDim.x * WAVES_PER_WG;

  int tile = gwave;
  if (tile < ntiles) prefetch_x_tile(x, tile, xlds[0], lane);

  int p = 0;
  for (; tile < ntiles; tile += nwaves) {
    const int nxt = tile + nwaves;
    if (nxt < ntiles) {
      prefetch_x_tile(x, nxt, xlds[p ^ 1], lane);           // overlap next tile
      asm volatile("s_wait_asynccnt 0x20" ::: "memory");    // current tile's 32 done
    } else {
      asm volatile("s_wait_asynccnt 0x0" ::: "memory");
    }
    const float* xrow = (p ? xbuf1 : xbuf0) + m * XS;

    // Layer 1: 256 -> 128  (64 WMMA)
    v8f a1[8];
    init_bias<8>(a1, b1, 128, lane);
    gemm_l1(xrow, wt + WT_L1, a1, lane);
    epilogue_relu<8>(a1, hbuf, lane);
    // Layer 2: 128 -> 64   (16 WMMA)
    v8f a2[4];
    init_bias<4>(a2, b2, 64, lane);
    gemm_bf16<4, 4>(hbuf, wt + WT_L2, a2, lane);
    epilogue_relu<4>(a2, hbuf, lane);
    // Layer 3: 64 -> 32    (4 WMMA)
    v8f a3[2];
    init_bias<2>(a3, b3, 32, lane);
    gemm_bf16<2, 2>(hbuf, wt + WT_L3, a3, lane);
    epilogue_relu<2>(a3, hbuf, lane);
    // Layer 4: 32 -> 16    (1 WMMA)
    v8f a4[1];
    init_bias<1>(a4, b4, 16, lane);
    gemm_bf16<1, 1>(hbuf, wt + WT_L4, a4, lane);
    epilogue_relu<1>(a4, hbuf, lane);
    // Layer 5: 16 -> 8     (1 WMMA, K/N zero-padded in B)
    v8f a5[1];
    init_bias<1>(a5, b5, 8, lane);
    gemm_bf16<1, 1>(hbuf, wt + WT_L5, a5, lane);
    epilogue_relu<1>(a5, hbuf, lane);
    // Layer 6: 8 -> 4
    v8f a6[1];
    init_bias<1>(a6, b6, 4, lane);
    gemm_bf16<1, 1>(hbuf, wt + WT_L6, a6, lane);
    epilogue_relu<1>(a6, hbuf, lane);
    // Layer 7: 4 -> 1, sigmoid, column 0 only
    v8f a7[1];
    init_bias<1>(a7, b7, 1, lane);
    gemm_bf16<1, 1>(hbuf, wt + WT_L7, a7, lane);
    if (m == 0) {  // lanes 0 and 16 hold column 0: rows 8h+0..8h+7
      float r[8];
#pragma unroll
      for (int v = 0; v < 8; ++v) {
        r[v] = 1.0f / (1.0f + __expf(-a7[0][v]));
      }
      float4* o = (float4*)(out + (size_t)tile * 16 + h * 8);
      o[0] = make_float4(r[0], r[1], r[2], r[3]);
      o[1] = make_float4(r[4], r[5], r[6], r[7]);
    }
    p ^= 1;
  }
}

// ---------------------------------------------------------------------------
extern "C" void kernel_launch(void* const* d_in, const int* in_sizes, int n_in,
                              void* d_out, int out_size, void* d_ws, size_t ws_size,
                              hipStream_t stream) {
  const float* x = (const float*)d_in[0];
  const float* W[7];
  const float* B[7];
  for (int i = 0; i < 7; ++i) {
    W[i] = (const float*)d_in[1 + 2 * i];
    B[i] = (const float*)d_in[2 + 2 * i];
  }
  unsigned short* wt = (unsigned short*)d_ws;  // needs WT_TOTAL*2 = 90112 B
  const int batch  = in_sizes[0] / 256;
  const int ntiles = batch / 16;

  prep_wtiles<<<88, 32, 0, stream>>>(W[0], W[1], W[2], W[3], W[4], W[5], W[6], wt);
  mlp_fused<<<NWG, WG_SIZE, SMEM_BYTES, stream>>>(
      x, wt, B[0], B[1], B[2], B[3], B[4], B[5], B[6], (float*)d_out, ntiles);
}
